// DualBranchGeometricEnhancer_11596411699694
// MI455X (gfx1250) — matrix operand order repaired and checked by reading, not compiled, not run on hardware
//
#include <hip/hip_runtime.h>
#include <hip/hip_bf16.h>

#define BN   4
#define NPTS 4096
#define KNNK 16
#define HDIM 128

typedef __attribute__((ext_vector_type(16))) __bf16 v16bf;
typedef __attribute__((ext_vector_type(8)))  __bf16 v8bf;
typedef __attribute__((ext_vector_type(8)))  float  v8f;

__device__ __forceinline__ float gelu_exact(float x) {
    return 0.5f * x * (1.0f + erff(x * 0.70710678118654752f));
}

// A-fragment (16x32 bf16, row-major source in LDS, pitch in elements).
// Lane L: m = L&15, half = L>>4. element e -> k = kbase + 16*half + e.
// Two contiguous 16B LDS loads.
__device__ __forceinline__ v16bf ld_frag_a(const __bf16* S, int pitch, int row0, int kbase) {
    const int lane = threadIdx.x & 31;
    const int half = lane >> 4;
    const int m    = lane & 15;
    const __bf16* p = S + (row0 + m) * pitch + kbase + half * 16;
    v8bf lo = *(const v8bf*)(p);
    v8bf hi = *(const v8bf*)(p + 8);
    v16bf r;
#pragma unroll
    for (int i = 0; i < 8; ++i) { r[i] = lo[i]; r[i + 8] = hi[i]; }
    return r;
}

// B-fragment (32x16 bf16) from a TRANSPOSED weight tile WT[n][k] (pitch elems).
// Lane L: n = L&15, half = L>>4, element e -> k = kbase + 16*half + e
// => 16 contiguous bf16 per lane.
__device__ __forceinline__ v16bf ld_frag_b(const __bf16* WT, int pitch, int col0, int kbase) {
    const int lane = threadIdx.x & 31;
    const int half = lane >> 4;
    const int n    = lane & 15;
    const __bf16* p = WT + (col0 + n) * pitch + kbase + half * 16;
    v8bf lo = *(const v8bf*)(p);
    v8bf hi = *(const v8bf*)(p + 8);
    v16bf r;
#pragma unroll
    for (int i = 0; i < 8; ++i) { r[i] = lo[i]; r[i + 8] = hi[i]; }
    return r;
}

__device__ __forceinline__ v8f wmma_bf16(v16bf a, v16bf b, v8f c) {
    return __builtin_amdgcn_wmma_f32_16x16x32_bf16(false, a, false, b, (short)0, c, false, false);
}

// ---------------------------------------------------------------------------
// KNN: brute force over LDS tiles (staged via gfx1250 async global->LDS copy),
// per-thread register insertion sort (k=16), prefetch of next tile.
// ---------------------------------------------------------------------------
__global__ __launch_bounds__(256) void knn_kernel(const float* __restrict__ points,
                                                  int* __restrict__ idx_out) {
    __shared__ __align__(16) float tile[256 * 3];
    const int b = blockIdx.y;
    const int q = blockIdx.x * 256 + threadIdx.x;
    const float* P = points + (size_t)b * NPTS * 6;
    const float qx = P[q * 6 + 0], qy = P[q * 6 + 1], qz = P[q * 6 + 2];
    // LDS byte address of this thread's tile slot (flat-LDS: low 32 bits).
    const unsigned lds_addr = (unsigned)(uintptr_t)(&tile[threadIdx.x * 3]);
    float bd[KNNK]; int bi[KNNK];
#pragma unroll
    for (int i = 0; i < KNNK; ++i) { bd[i] = 3.0e38f; bi[i] = 0; }
    for (int t = 0; t < NPTS / 256; ++t) {
        const int cbase = t * 256;
        __syncthreads();  // previous tile fully consumed
        {
            const int c = cbase + threadIdx.x;
            const unsigned long long gaddr =
                (unsigned long long)(uintptr_t)(P + (size_t)c * 6);
            // async copy 3 dwords (xyz); offset: is added to BOTH addresses
            asm volatile(
                "global_load_async_to_lds_b32 %0, %1, off offset:0\n\t"
                "global_load_async_to_lds_b32 %0, %1, off offset:4\n\t"
                "global_load_async_to_lds_b32 %0, %1, off offset:8"
                :: "v"(lds_addr), "v"(gaddr) : "memory");
            asm volatile("s_wait_asynccnt 0x0" ::: "memory");
        }
        __syncthreads();
        if (t + 1 < NPTS / 256) {  // overlap next tile's HBM fetch with compute
            __builtin_prefetch(P + (size_t)(cbase + 256 + threadIdx.x) * 6, 0, 1);
        }
#pragma unroll 1
        for (int cc = 0; cc < 256; ++cc) {
            const int cand = cbase + cc;
            const float dx = tile[cc * 3 + 0] - qx;
            const float dy = tile[cc * 3 + 1] - qy;
            const float dz = tile[cc * 3 + 2] - qz;
            const float d2 = dx * dx + dy * dy + dz * dz;
            if (cand == q) continue;
            if (d2 < bd[KNNK - 1]) {
                float d = d2; int id = cand;
#pragma unroll
                for (int i = 0; i < KNNK; ++i) {
                    if (d < bd[i]) {
                        float td = bd[i]; bd[i] = d;  d  = td;
                        int   ti = bi[i]; bi[i] = id; id = ti;
                    }
                }
            }
        }
    }
    int* o = idx_out + ((size_t)b * NPTS + q) * KNNK;
#pragma unroll
    for (int i = 0; i < KNNK; ++i) o[i] = bi[i];
}

// ---------------------------------------------------------------------------
// Branch MLP: 8 points x 16 neighbors per block (128 rows).
// layer1 (3/4 -> 128) VALU+gelu -> LDS bf16; layer2 (128x128) WMMA; max over K.
// MODE 0 = coord branch (rel_xyz), MODE 1 = normal branch (neigh_n, variation).
// ---------------------------------------------------------------------------
template <int MODE>
__global__ __launch_bounds__(256) void branch_kernel(
        const float* __restrict__ points, const int* __restrict__ idx,
        const float* __restrict__ W1, const float* __restrict__ b1,
        const float* __restrict__ W2, const float* __restrict__ b2,
        float* __restrict__ feat) {
    __shared__ __align__(16) __bf16 hbuf[128 * HDIM];   // 32 KB: gelu(layer1)
    __shared__ __align__(16) __bf16 wbuf[HDIM * HDIM];  // 32 KB: W2^T [n][k]
    const int b   = blockIdx.y;
    const int p0  = blockIdx.x * 8;
    const int tid = threadIdx.x;

    // layer 1: two threads per row, 64 output columns each
    {
        const int row = tid & 127;
        const int ch  = tid >> 7;
        const int pl  = row >> 4;     // local point 0..7
        const int j   = row & 15;     // neighbor slot
        const size_t base = (size_t)b * NPTS + (p0 + pl);
        const int nb = idx[base * KNNK + j];
        const float* pc = points + base * 6;
        const float* pn = points + ((size_t)b * NPTS + nb) * 6;
        float in0, in1, in2, in3 = 0.0f;
        if (MODE == 0) {
            in0 = pn[0] - pc[0]; in1 = pn[1] - pc[1]; in2 = pn[2] - pc[2];
        } else {
            in0 = pn[3]; in1 = pn[4]; in2 = pn[5];
            in3 = 1.0f - (pc[3] * in0 + pc[4] * in1 + pc[5] * in2);
        }
        for (int c = 0; c < 64; ++c) {
            const int col = ch * 64 + c;
            float acc = b1[col];
            acc += in0 * W1[0 * HDIM + col];
            acc += in1 * W1[1 * HDIM + col];
            acc += in2 * W1[2 * HDIM + col];
            if (MODE == 1) acc += in3 * W1[3 * HDIM + col];
            hbuf[row * HDIM + col] = (__bf16)gelu_exact(acc);
        }
    }
    // stage W2^T as bf16
    for (int i = 0; i < 64; ++i) {
        const int e = i * 256 + tid;
        const int k = e >> 7, n = e & 127;
        wbuf[n * HDIM + k] = (__bf16)W2[k * HDIM + n];
    }
    __syncthreads();

    // layer 2: wave w owns point w (one 16-row M-tile), 8 N-tiles, K=128
    const int wv   = tid >> 5;
    const int lane = tid & 31;
    v16bf afrag[4];
#pragma unroll
    for (int ks = 0; ks < 4; ++ks) afrag[ks] = ld_frag_a(hbuf, HDIM, wv * 16, ks * 32);
#pragma unroll
    for (int nt = 0; nt < 8; ++nt) {
        // issue all four B-fragment loads before the WMMA chain so the
        // matrix pipe overlaps the LDS latency (partial dscnt waits)
        v16bf bb0 = ld_frag_b(wbuf, HDIM, nt * 16, 0);
        v16bf bb1 = ld_frag_b(wbuf, HDIM, nt * 16, 32);
        v16bf bb2 = ld_frag_b(wbuf, HDIM, nt * 16, 64);
        v16bf bb3 = ld_frag_b(wbuf, HDIM, nt * 16, 96);
        v8f acc = {};
        acc = wmma_bf16(afrag[0], bb0, acc);
        acc = wmma_bf16(afrag[1], bb1, acc);
        acc = wmma_bf16(afrag[2], bb2, acc);
        acc = wmma_bf16(afrag[3], bb3, acc);
        // max over the 16 neighbors = max over the M-tile rows
        float mx = acc[0];
#pragma unroll
        for (int r = 1; r < 8; ++r) mx = fmaxf(mx, acc[r]);
        mx = fmaxf(mx, __shfl_xor(mx, 16, 32));
        if (lane < 16) {
            const int col = nt * 16 + lane;
            feat[((size_t)b * NPTS + p0 + wv) * HDIM + col] = mx + b2[col];
        }
    }
}

// ---------------------------------------------------------------------------
// Head MLP: 64 rows per block. GEMM1 (256->128, W1^T double-chunked in LDS),
// gelu -> bf16 H (reuses xbuf), GEMM2 (128->128), bias, fp32 store.
// ---------------------------------------------------------------------------
__global__ __launch_bounds__(256) void head_kernel(
        const float* __restrict__ cf, const float* __restrict__ nf,
        const float* __restrict__ Wo1, const float* __restrict__ bo1,
        const float* __restrict__ Wo2, const float* __restrict__ bo2,
        float* __restrict__ out) {
    __shared__ __align__(16) __bf16 xbuf[64 * 256];   // 32 KB: X, then H (pitch 128)
    __shared__ __align__(16) __bf16 wbuf[128 * 128];  // 32 KB: W^T chunk
    const int tid = threadIdx.x;
    const size_t g0 = (size_t)blockIdx.x * 64;

    // stage X = [coord_feat | normal_feat] as bf16
    for (int i = 0; i < 64; ++i) {
        const int e = i * 256 + tid;
        const int r = e >> 8, c = e & 255;
        const float v = (c < 128) ? cf[(g0 + r) * 128 + c]
                                  : nf[(g0 + r) * 128 + (c - 128)];
        xbuf[r * 256 + c] = (__bf16)v;
    }
    const int wv   = tid >> 5;
    const int lane = tid & 31;
    const int mt   = wv >> 1;         // 4 M-tiles
    const int ntb  = (wv & 1) * 4;    // 4 N-tiles per wave

    v8f acc[4] = {};
    for (int kt = 0; kt < 2; ++kt) {
        __syncthreads();
        for (int i = 0; i < 64; ++i) {
            const int e = i * 256 + tid;
            const int kk = e >> 7, n = e & 127;
            wbuf[n * 128 + kk] = (__bf16)Wo1[((size_t)(kt * 128 + kk)) * 128 + n];
        }
        __syncthreads();
#pragma unroll
        for (int ks = 0; ks < 4; ++ks) {
            v16bf a = ld_frag_a(xbuf, 256, mt * 16, kt * 128 + ks * 32);
            v16bf bb0 = ld_frag_b(wbuf, 128, (ntb + 0) * 16, ks * 32);
            v16bf bb1 = ld_frag_b(wbuf, 128, (ntb + 1) * 16, ks * 32);
            v16bf bb2 = ld_frag_b(wbuf, 128, (ntb + 2) * 16, ks * 32);
            v16bf bb3 = ld_frag_b(wbuf, 128, (ntb + 3) * 16, ks * 32);
            acc[0] = wmma_bf16(a, bb0, acc[0]);
            acc[1] = wmma_bf16(a, bb1, acc[1]);
            acc[2] = wmma_bf16(a, bb2, acc[2]);
            acc[3] = wmma_bf16(a, bb3, acc[3]);
        }
    }
    __syncthreads();
    // H = gelu(acc + bo1) -> xbuf (pitch 128, X fully consumed); W2^T -> wbuf
#pragma unroll
    for (int j = 0; j < 4; ++j) {
        const int col  = (ntb + j) * 16 + (lane & 15);
        const float bv = bo1[col];
#pragma unroll
        for (int r = 0; r < 8; ++r) {
            const int m = mt * 16 + r + 8 * (lane >> 4);
            xbuf[m * 128 + col] = (__bf16)gelu_exact(acc[j][r] + bv);
        }
    }
    for (int i = 0; i < 64; ++i) {
        const int e = i * 256 + tid;
        const int kk = e >> 7, n = e & 127;
        wbuf[n * 128 + kk] = (__bf16)Wo2[(size_t)kk * 128 + n];
    }
    __syncthreads();

    v8f acc2[4] = {};
#pragma unroll
    for (int ks = 0; ks < 4; ++ks) {
        v16bf a = ld_frag_a(xbuf, 128, mt * 16, ks * 32);
        v16bf bb0 = ld_frag_b(wbuf, 128, (ntb + 0) * 16, ks * 32);
        v16bf bb1 = ld_frag_b(wbuf, 128, (ntb + 1) * 16, ks * 32);
        v16bf bb2 = ld_frag_b(wbuf, 128, (ntb + 2) * 16, ks * 32);
        v16bf bb3 = ld_frag_b(wbuf, 128, (ntb + 3) * 16, ks * 32);
        acc2[0] = wmma_bf16(a, bb0, acc2[0]);
        acc2[1] = wmma_bf16(a, bb1, acc2[1]);
        acc2[2] = wmma_bf16(a, bb2, acc2[2]);
        acc2[3] = wmma_bf16(a, bb3, acc2[3]);
    }
#pragma unroll
    for (int j = 0; j < 4; ++j) {
        const int col  = (ntb + j) * 16 + (lane & 15);
        const float bv = bo2[col];
#pragma unroll
        for (int r = 0; r < 8; ++r) {
            const size_t m = g0 + mt * 16 + r + 8 * (lane >> 4);
            out[m * 128 + col] = acc2[j][r] + bv;
        }
    }
}

// ---------------------------------------------------------------------------
extern "C" void kernel_launch(void* const* d_in, const int* in_sizes, int n_in,
                              void* d_out, int out_size, void* d_ws, size_t ws_size,
                              hipStream_t stream) {
    (void)in_sizes; (void)n_in; (void)out_size; (void)ws_size;
    const float* points = (const float*)d_in[0];
    const float* Wc1 = (const float*)d_in[1];
    const float* bc1 = (const float*)d_in[2];
    const float* Wc2 = (const float*)d_in[3];
    const float* bc2 = (const float*)d_in[4];
    const float* Wn1 = (const float*)d_in[5];
    const float* bn1 = (const float*)d_in[6];
    const float* Wn2 = (const float*)d_in[7];
    const float* bn2 = (const float*)d_in[8];
    const float* Wo1 = (const float*)d_in[9];
    const float* bo1 = (const float*)d_in[10];
    const float* Wo2 = (const float*)d_in[11];
    const float* bo2 = (const float*)d_in[12];
    float* out = (float*)d_out;

    char* ws = (char*)d_ws;
    int*   idx   = (int*)ws;
    float* cfeat = (float*)(ws + (size_t)BN * NPTS * KNNK * sizeof(int));
    float* nfeat = cfeat + (size_t)BN * NPTS * HDIM;

    knn_kernel<<<dim3(NPTS / 256, BN), 256, 0, stream>>>(points, idx);
    branch_kernel<0><<<dim3(NPTS / 8, BN), 256, 0, stream>>>(points, idx, Wc1, bc1, Wc2, bc2, cfeat);
    branch_kernel<1><<<dim3(NPTS / 8, BN), 256, 0, stream>>>(points, idx, Wn1, bn1, Wn2, bn2, nfeat);
    head_kernel<<<(BN * NPTS) / 64, 256, 0, stream>>>(cfeat, nfeat, Wo1, bo1, Wo2, bo2, out);
}
